// SynapseNet_GAT_MLP_4037269258378
// MI455X (gfx1250) — compile-verified, matrix operand order checked by reading, not run on hardware
//
#include <hip/hip_runtime.h>
#include <hip/hip_bf16.h>

// ---------------------------------------------------------------------------
// SynapseNet (GAT + MLP) forward for gfx1250 (MI455X, wave32, WMMA).
// All GEMM operands are padded to 64x64x32 tile multiples in workspace, so
// the WMMA GEMM is a single guard-free path: bulk global loads -> LDS tiles
// staged in the WMMA fragment layout (ISA 7.12.2) -> 2x ds_load_b128 per
// fragment -> 4x v_wmma_f32_16x16x32_f16 per wave per k-step.
// Scatter/softmax phases use fp32 global atomics.
// ---------------------------------------------------------------------------

typedef _Float16 half_t;
typedef __attribute__((ext_vector_type(16))) _Float16 v16h;
typedef __attribute__((ext_vector_type(8)))  _Float16 v8h;
typedef __attribute__((ext_vector_type(8)))  float    v8f;
typedef __attribute__((ext_vector_type(4)))  float    v4f;

#define TM 64
#define TN 64
#define TK 32
#define PSTR 32   // padded row stride of the point-feature matrix (K=6 -> 32)

__device__ __forceinline__ void atomic_add_f(float* p, float v) {
  __hip_atomic_fetch_add(p, v, __ATOMIC_RELAXED, __HIP_MEMORY_SCOPE_AGENT);
}
__device__ __forceinline__ void atomic_max_u(unsigned* p, unsigned v) {
  __hip_atomic_fetch_max(p, v, __ATOMIC_RELAXED, __HIP_MEMORY_SCOPE_AGENT);
}
// order-preserving float <-> uint encoding (for atomic float max)
__device__ __forceinline__ unsigned enc_f32(float f) {
  unsigned u = __float_as_uint(f);
  return (u & 0x80000000u) ? ~u : (u | 0x80000000u);
}
__device__ __forceinline__ float dec_f32(unsigned u) {
  return __uint_as_float((u & 0x80000000u) ? (u & 0x7FFFFFFFu) : ~u);
}
#define ENC_NEG_FLT_MAX 0x00800000u  // enc(-FLT_MAX)

// ---------------------------------------------------------------------------
// Tile-exact GEMM: C = act(accum*C + A[M,K] @ B[K,N] + bias)
// Preconditions (guaranteed by driver): M%64==0, N%64==0, K%32==0.
// Block = 128 threads = 4 waves; block tile 64x64; wave tile 32x32 (2x2 WMMA).
// LDS layout = WMMA fragment layout: slot[subtile][lane][0..15] holds the 16
// halves lane needs, in element order k = {0..7,16..23} + 8*(lane>>4).
// act: 0 none, 1 relu, 2 elu.
// ---------------------------------------------------------------------------
__global__ __launch_bounds__(128)
void gemm_wmma(const float* __restrict__ A, const float* __restrict__ B,
               const float* __restrict__ bias, float* __restrict__ C,
               int Nc, int K, int act, int accum)
{
  __shared__ alignas(32) half_t As[4][32][16];   // [row subtile][lane][elem]
  __shared__ alignas(32) half_t Bs[4][32][16];   // [col subtile][lane][elem]

  const int tid    = threadIdx.x;
  const int lane   = tid & 31;
  const int wave   = tid >> 5;
  const int wr     = (wave >> 1) * 32;   // wave row offset in block tile
  const int wc     = (wave & 1) * 32;    // wave col offset in block tile
  const int r0     = blockIdx.y * TM;
  const int c0     = blockIdx.x * TN;
  const int half16 = lane >> 4;

  // staging assignment: thread owns (row|col sm, k-half hsel) -> one LDS slot
  const int sm    = tid & 63;
  const int hsel  = tid >> 6;
  const int dsub  = sm >> 4;
  const int dlane = (sm & 15) + 16 * hsel;
  half_t* aslot = &As[dsub][dlane][0];
  half_t* bslot = &Bs[dsub][dlane][0];

  const float* aptr = A + (size_t)(r0 + sm) * K;
  const float* bptr = B + (c0 + sm);

  v8f acc[2][2] = {};

  for (int k0 = 0; k0 < K; k0 += TK) {
    // ---- stage A slot: 4 x b128 loads, convert, 2 x ds_store_b128 --------
    {
      alignas(16) half_t ta[16];
#pragma unroll
      for (int g = 0; g < 4; ++g) {
        int kb = (g & 1) * 4 + (g >> 1) * 16 + 8 * hsel;  // 0,4,16,20 (+8)
        v4f v = *(const v4f*)(aptr + k0 + kb);
#pragma unroll
        for (int i = 0; i < 4; ++i) ta[g * 4 + i] = (half_t)v[i];
      }
      *(v8h*)(aslot)     = *(const v8h*)&ta[0];
      *(v8h*)(aslot + 8) = *(const v8h*)&ta[8];
    }
    // ---- stage B slot: 16 lane-coalesced b32 loads, 2 x ds_store_b128 ----
    {
      alignas(16) half_t tb[16];
#pragma unroll
      for (int j = 0; j < 16; ++j) {
        int k = (j & 7) + (j >> 3) * 16 + 8 * hsel;
        tb[j] = (half_t)bptr[(size_t)(k0 + k) * Nc];
      }
      *(v8h*)(bslot)     = *(const v8h*)&tb[0];
      *(v8h*)(bslot + 8) = *(const v8h*)&tb[8];
    }
    __syncthreads();

    // ---- fragments: two ds_load_b128 each, zero repacking -----------------
    v16h af0 = *(const v16h*)As[(wr >> 4) + 0][lane];
    v16h af1 = *(const v16h*)As[(wr >> 4) + 1][lane];
    v16h bf0 = *(const v16h*)Bs[(wc >> 4) + 0][lane];
    v16h bf1 = *(const v16h*)Bs[(wc >> 4) + 1][lane];

    acc[0][0] = __builtin_amdgcn_wmma_f32_16x16x32_f16(false, af0, false, bf0,
                                                       (short)0, acc[0][0], false, false);
    acc[0][1] = __builtin_amdgcn_wmma_f32_16x16x32_f16(false, af0, false, bf1,
                                                       (short)0, acc[0][1], false, false);
    acc[1][0] = __builtin_amdgcn_wmma_f32_16x16x32_f16(false, af1, false, bf0,
                                                       (short)0, acc[1][0], false, false);
    acc[1][1] = __builtin_amdgcn_wmma_f32_16x16x32_f16(false, af1, false, bf1,
                                                       (short)0, acc[1][1], false, false);
    __syncthreads();
  }

  // Epilogue: C/D layout — lane col = lane&15, VGPR r -> row r + 8*(lane>>4)
  const int l16 = lane & 15;
#pragma unroll
  for (int i = 0; i < 2; ++i) {
#pragma unroll
    for (int j = 0; j < 2; ++j) {
#pragma unroll
      for (int r = 0; r < 8; ++r) {
        int gr = r0 + wr + i * 16 + r + 8 * half16;
        int gc = c0 + wc + j * 16 + l16;
        float v = acc[i][j][r];
        if (accum) v += C[(size_t)gr * Nc + gc];
        if (bias)  v += bias[gc];
        if (act == 1)      v = fmaxf(v, 0.f);
        else if (act == 2) v = (v > 0.f) ? v : (expf(v) - 1.f);
        C[(size_t)gr * Nc + gc] = v;
      }
    }
  }
}

// ---------------------------------------------------------------------------
// Utility / graph kernels
// ---------------------------------------------------------------------------
__global__ __launch_bounds__(256)
void fill_f32(float* p, float v, long n) {
  long i = (long)blockIdx.x * blockDim.x + threadIdx.x;
  if (i < n) p[i] = v;
}
__global__ __launch_bounds__(256)
void fill_u32_strided(unsigned* p, unsigned v, long rows, int cols, int stride) {
  long i = (long)blockIdx.x * blockDim.x + threadIdx.x;
  if (i >= rows * cols) return;
  long r = i / cols; int c = (int)(i % cols);
  p[r * stride + c] = v;
}
// strided 2D copy (also used to (un)pad weight/output matrices)
__global__ __launch_bounds__(256)
void copy2d(float* __restrict__ dst, int dstride,
            const float* __restrict__ src, int sstride, long rows, int cols) {
  long i = (long)blockIdx.x * blockDim.x + threadIdx.x;
  if (i >= rows * cols) return;
  long r = i / cols; int c = (int)(i % cols);
  dst[r * dstride + c] = src[r * sstride + c];
}

// per-(node,head) attention logits: s = <xl[n,h,:], a_src[h,:]>, d likewise
__global__ __launch_bounds__(256)
void node_sd(const float* __restrict__ xl, const float* __restrict__ asrc,
             const float* __restrict__ adst, float* __restrict__ s,
             float* __restrict__ d, int n, int Hh, int Cc)
{
  int i = blockIdx.x * blockDim.x + threadIdx.x;
  if (i >= n * Hh) return;
  int node = i / Hh, h = i % Hh;
  int stride = Hh * Cc;
  const float* row = xl + (size_t)node * stride + h * Cc;
  const float* va = asrc + h * Cc;
  const float* vd = adst + h * Cc;
  float ss = 0.f, dd = 0.f;
  for (int c = 0; c < Cc; ++c) { ss += row[c] * va[c]; dd += row[c] * vd[c]; }
  s[i] = ss; d[i] = dd;
}

// pass 1: alpha = leaky_relu(s[src]+d[dst]); segment-max over dst (encoded)
__global__ __launch_bounds__(256)
void edge_alpha_max(const int* __restrict__ ei, const float* __restrict__ s,
                    const float* __restrict__ d, float* __restrict__ alpha,
                    unsigned* __restrict__ amax, int Eg, int Nn, int Hh)
{
  int e = blockIdx.x * blockDim.x + threadIdx.x;
  if (e >= Eg + Nn) return;
  int sn, dn;
  if (e < Eg) { sn = ei[e]; dn = ei[Eg + e]; } else { sn = dn = e - Eg; }
  for (int h = 0; h < Hh; ++h) {
    float a = s[sn * Hh + h] + d[dn * Hh + h];
    a = (a > 0.f) ? a : 0.2f * a;
    alpha[(size_t)e * Hh + h] = a;
    atomic_max_u(&amax[dn * Hh + h], enc_f32(a));
  }
}

// pass 2: ex = exp(alpha - amax[dst]); den[dst] += ex  (alpha overwritten)
__global__ __launch_bounds__(256)
void edge_exp_den(const int* __restrict__ ei, float* __restrict__ alpha,
                  const unsigned* __restrict__ amax, float* __restrict__ den,
                  int Eg, int Nn, int Hh)
{
  int e = blockIdx.x * blockDim.x + threadIdx.x;
  if (e >= Eg + Nn) return;
  int dn = (e < Eg) ? ei[Eg + e] : (e - Eg);
  for (int h = 0; h < Hh; ++h) {
    float ex = expf(alpha[(size_t)e * Hh + h] - dec_f32(amax[dn * Hh + h]));
    alpha[(size_t)e * Hh + h] = ex;
    atomic_add_f(&den[dn * Hh + h], ex);
  }
}

// pass 3: out[dst,ch] += xl[src,ch] * ex/(den[dst]+1e-16)
__global__ __launch_bounds__(256)
void edge_aggregate(const int* __restrict__ ei, const float* __restrict__ xl,
                    const float* __restrict__ ex, const float* __restrict__ den,
                    float* __restrict__ out, int Eg, int Nn, int Hh, int Cc)
{
  int stride = Hh * Cc;
  long idx = (long)blockIdx.x * blockDim.x + threadIdx.x;
  long total = (long)(Eg + Nn) * stride;
  if (idx >= total) return;
  int e  = (int)(idx / stride);
  int ch = (int)(idx % stride);
  int h  = ch / Cc;
  int sn, dn;
  if (e < Eg) { sn = ei[e]; dn = ei[Eg + e]; } else { sn = dn = e - Eg; }
  float w = ex[(size_t)e * Hh + h] / (den[dn * Hh + h] + 1e-16f);
  atomic_add_f(&out[(size_t)dn * stride + ch], xl[(size_t)sn * stride + ch] * w);
}

// x[n,ch] = act(x[n,ch] + b[ch])
__global__ __launch_bounds__(256)
void bias_act(float* __restrict__ x, const float* __restrict__ b,
              int n, int ch, int act)
{
  long i = (long)blockIdx.x * blockDim.x + threadIdx.x;
  if (i >= (long)n * ch) return;
  float v = x[i] + b[(int)(i % ch)];
  if (act == 1)      v = fmaxf(v, 0.f);
  else if (act == 2) v = (v > 0.f) ? v : (expf(v) - 1.f);
  x[i] = v;
}

// segment max over synapse points (encoded; rows are PSTR-padded)
__global__ __launch_bounds__(256)
void point_max(const float* __restrict__ syn, const int* __restrict__ sidx,
               unsigned* __restrict__ encmax, int Pp)
{
  int p = blockIdx.x * blockDim.x + threadIdx.x;
  if (p >= Pp) return;
  int seg = sidx[p];
  for (int c = 0; c < 6; ++c)
    atomic_max_u(&encmax[(size_t)seg * PSTR + c], enc_f32(syn[(size_t)p * 6 + c]));
}
__global__ __launch_bounds__(256)
void decode_max(unsigned* __restrict__ p, int rows) {
  long i = (long)blockIdx.x * blockDim.x + threadIdx.x;
  if (i >= (long)rows * 6) return;
  long r = i / 6; int c = (int)(i % 6);
  float v = dec_f32(p[r * PSTR + c]);
  ((float*)p)[r * PSTR + c] = v;
}

__global__ __launch_bounds__(256)
void edge_counts(const int* __restrict__ ei, float* __restrict__ cl,
                 float* __restrict__ cr, int Eg)
{
  int e = blockIdx.x * blockDim.x + threadIdx.x;
  if (e >= Eg) return;
  atomic_add_f(&cl[ei[e]], 1.f);
  atomic_add_f(&cr[ei[Eg + e]], 1.f);
}

// scatter xp chunk into left/right sums
__global__ __launch_bounds__(256)
void scatter_xp(const int* __restrict__ ei, const float* __restrict__ xp,
                float* __restrict__ left, float* __restrict__ right,
                int est, int ecnt, int Eg)
{
  long i = (long)blockIdx.x * blockDim.x + threadIdx.x;
  if (i >= (long)ecnt * 512) return;
  int el = (int)(i / 512), ch = (int)(i % 512);
  int eg = est + el;
  if (eg >= Eg) return;
  float v = xp[(size_t)el * 512 + ch];
  atomic_add_f(&left [(size_t)ei[eg]      * 512 + ch], v);
  atomic_add_f(&right[(size_t)ei[Eg + eg] * 512 + ch], v);
}

__global__ __launch_bounds__(256)
void div_counts(float* __restrict__ x, const float* __restrict__ cnt,
                int n, int ch)
{
  long i = (long)blockIdx.x * blockDim.x + threadIdx.x;
  if (i >= (long)n * ch) return;
  x[i] /= fmaxf(cnt[(int)(i / ch)], 1.f);
}

// ---------------------------------------------------------------------------
// Host driver
// ---------------------------------------------------------------------------
static inline int cdiv(long a, long b) { return (int)((a + b - 1) / b); }

extern "C" void kernel_launch(void* const* d_in, const int* in_sizes, int n_in,
                              void* d_out, int out_size, void* d_ws, size_t ws_size,
                              hipStream_t stream)
{
  const int*   ei      = (const int*)d_in[0];
  const float* synapse = (const float*)d_in[2];
  const int*   sidx    = (const int*)d_in[3];
  const float* x_param = (const float*)d_in[5];
  const float* W1  = (const float*)d_in[6];
  const float* as1 = (const float*)d_in[7];
  const float* ad1 = (const float*)d_in[8];
  const float* b1  = (const float*)d_in[9];
  const float* W2  = (const float*)d_in[10];
  const float* as2 = (const float*)d_in[11];
  const float* ad2 = (const float*)d_in[12];
  const float* b2  = (const float*)d_in[13];
  const float* We1 = (const float*)d_in[14];
  const float* be1 = (const float*)d_in[15];
  const float* We2 = (const float*)d_in[16];
  const float* be2 = (const float*)d_in[17];
  const float* Wc1 = (const float*)d_in[18];
  const float* bc1 = (const float*)d_in[19];
  const float* Wc2 = (const float*)d_in[20];
  const float* bc2 = (const float*)d_in[21];

  const int Nn  = in_sizes[5] / 128;    // 20000 nodes
  const int Eg  = in_sizes[0] / 2;      // 200000 edges (multiple of 64)
  const int Pp  = in_sizes[2] / 6;      // 800000 points
  const int NC  = in_sizes[21];         // 133 classes
  const int EL  = Eg + Nn;              // edges incl. self loops
  const int Mp  = cdiv(Nn, TM) * TM;    // node rows padded to 64  (20032)
  const int NCp = cdiv(NC, TN) * TN;    // class cols padded to 64 (192)
  const int CH  = 12800;                // point-MLP chunk (multiple of 64)

  // workspace layout (floats); all GEMM operands tile-exact
  float* W = (float*)d_ws;
  size_t o = 0;
  float*    xl    = W + o; o += (size_t)Mp * 512;  // xl1 / xt2 / c1
  float*    xout  = W + o; o += (size_t)Mp * 512;  // out1->x1 / out2->x2
  float*    xpad  = W + o; o += (size_t)Mp * 128;  // padded copy of x_param
  float*    sbuf  = W + o; o += (size_t)Nn * 8;
  float*    dbuf  = W + o; o += (size_t)Nn * 8;
  float*    exb   = W + o; o += (size_t)EL * 8;
  unsigned* amax  = (unsigned*)(W + o); o += (size_t)Nn * 8;
  float*    den   = W + o; o += (size_t)Nn * 8;
  float*    pmaxp = W + o; o += (size_t)Eg * PSTR; // padded point features
  float*    we1p  = W + o; o += (size_t)PSTR * 256;// padded We1 (K 6->32)
  float*    hbuf  = W + o; o += (size_t)CH * 256;
  float*    xpb   = W + o; o += (size_t)CH * 512;
  float*    left  = W + o; o += (size_t)Mp * 512;
  float*    right = W + o; o += (size_t)Mp * 512;
  float*    cl    = W + o; o += (size_t)Nn;
  float*    cr    = W + o; o += (size_t)Nn;
  float*    wc2p  = W + o; o += (size_t)512 * NCp; // padded Wc2 (N 133->192)
  float*    bc2p  = W + o; o += (size_t)NCp;
  float*    outp  = W + o; o += (size_t)Mp * NCp;  // padded logits

  #define GEMM(A_, B_, bias_, C_, M_, N_, K_, act_, accum_)                 \
    gemm_wmma<<<dim3((N_) / TN, (M_) / TM), 128, 0, stream>>>(              \
        (A_), (B_), (bias_), (C_), (N_), (K_), (act_), (accum_))

  auto attn = [&](const float* xlp, const float* asrc, const float* adst,
                  float* outbuf, int Hh, int Cc) {
    int stride = Hh * Cc;
    fill_u32_strided<<<cdiv(Nn * Hh, 256), 256, 0, stream>>>(amax, ENC_NEG_FLT_MAX, Nn * Hh, 1, 1);
    fill_f32<<<cdiv(Nn * Hh, 256), 256, 0, stream>>>(den, 0.f, Nn * Hh);
    node_sd<<<cdiv(Nn * Hh, 256), 256, 0, stream>>>(xlp, asrc, adst, sbuf, dbuf, Nn, Hh, Cc);
    edge_alpha_max<<<cdiv(EL, 256), 256, 0, stream>>>(ei, sbuf, dbuf, exb, amax, Eg, Nn, Hh);
    edge_exp_den<<<cdiv(EL, 256), 256, 0, stream>>>(ei, exb, amax, den, Eg, Nn, Hh);
    edge_aggregate<<<cdiv((long)EL * stride, 256), 256, 0, stream>>>(
        ei, xlp, exb, den, outbuf, Eg, Nn, Hh, Cc);
  };

  // ---- pad x_param into xpad (pad rows never feed real output rows) ------
  copy2d<<<cdiv((long)Nn * 128, 256), 256, 0, stream>>>(xpad, 128, x_param, 128, Nn, 128);

  // ---- GAT layer 1: x1 = elu(GAT(x_param) + b1) --------------------------
  GEMM(xpad, W1, nullptr, xl, Mp, 512, 128, 0, 0);                 // xl1
  fill_f32<<<cdiv((long)Mp * 512, 256), 256, 0, stream>>>(xout, 0.f, (long)Mp * 512);
  attn(xl, as1, ad1, xout, 8, 64);
  bias_act<<<cdiv((long)Nn * 512, 256), 256, 0, stream>>>(xout, b1, Nn, 512, 2);

  // ---- GAT layer 2: x2 = GAT(x1) + b2 ------------------------------------
  GEMM(xout, W2, nullptr, xl, Mp, 512, 512, 0, 0);                 // xt2 (reuse xl)
  fill_f32<<<cdiv((long)Mp * 512, 256), 256, 0, stream>>>(xout, 0.f, (long)Mp * 512);
  attn(xl, as2, ad2, xout, 1, 512);
  bias_act<<<cdiv((long)Nn * 512, 256), 256, 0, stream>>>(xout, b2, Nn, 512, 0);
  // x2 lives in xout

  // ---- point branch: segment-max -> MLP -> scatter-mean ------------------
  fill_f32<<<cdiv((long)Eg * PSTR, 256), 256, 0, stream>>>(pmaxp, 0.f, (long)Eg * PSTR);
  fill_u32_strided<<<cdiv((long)Eg * 6, 256), 256, 0, stream>>>(
      (unsigned*)pmaxp, ENC_NEG_FLT_MAX, Eg, 6, PSTR);
  point_max<<<cdiv(Pp, 256), 256, 0, stream>>>(synapse, sidx, (unsigned*)pmaxp, Pp);
  decode_max<<<cdiv((long)Eg * 6, 256), 256, 0, stream>>>((unsigned*)pmaxp, Eg);

  fill_f32<<<cdiv((long)PSTR * 256, 256), 256, 0, stream>>>(we1p, 0.f, (long)PSTR * 256);
  copy2d<<<cdiv((long)6 * 256, 256), 256, 0, stream>>>(we1p, 256, We1, 256, 6, 256);

  fill_f32<<<cdiv((long)Mp * 512, 256), 256, 0, stream>>>(left, 0.f, (long)Mp * 512);
  fill_f32<<<cdiv((long)Mp * 512, 256), 256, 0, stream>>>(right, 0.f, (long)Mp * 512);
  fill_f32<<<cdiv(Nn, 256), 256, 0, stream>>>(cl, 0.f, Nn);
  fill_f32<<<cdiv(Nn, 256), 256, 0, stream>>>(cr, 0.f, Nn);
  edge_counts<<<cdiv(Eg, 256), 256, 0, stream>>>(ei, cl, cr, Eg);

  for (int est = 0; est < Eg; est += CH) {
    int ecnt = (Eg - est < CH) ? (Eg - est) : CH;   // always multiple of 64
    GEMM(pmaxp + (size_t)est * PSTR, we1p, be1, hbuf, ecnt, 256, PSTR, 1, 0);
    GEMM(hbuf, We2, be2, xpb, ecnt, 512, 256, 0, 0);
    scatter_xp<<<cdiv((long)ecnt * 512, 256), 256, 0, stream>>>(
        ei, xpb, left, right, est, ecnt, Eg);
  }
  div_counts<<<cdiv((long)Nn * 512, 256), 256, 0, stream>>>(left, cl, Nn, 512);
  div_counts<<<cdiv((long)Nn * 512, 256), 256, 0, stream>>>(right, cr, Nn, 512);

  // ---- classifier: pred = relu(concat(x2,left,right)@Wc1+bc1)@Wc2+bc2 ----
  // concat never materialized: three accumulating GEMMs over Wc1 row-blocks.
  GEMM(xout,  Wc1,               nullptr, xl, Mp, 512, 512, 0, 0);
  GEMM(left,  Wc1 + 512  * 512,  nullptr, xl, Mp, 512, 512, 0, 1);
  GEMM(right, Wc1 + 1024 * 512,  bc1,     xl, Mp, 512, 512, 1, 1);

  // pad Wc2/bc2 to NCp cols, GEMM into padded logits, then unpad into d_out
  fill_f32<<<cdiv((long)512 * NCp, 256), 256, 0, stream>>>(wc2p, 0.f, (long)512 * NCp);
  copy2d<<<cdiv((long)512 * NC, 256), 256, 0, stream>>>(wc2p, NCp, Wc2, NC, 512, NC);
  fill_f32<<<cdiv(NCp, 256), 256, 0, stream>>>(bc2p, 0.f, NCp);
  copy2d<<<cdiv(NC, 256), 256, 0, stream>>>(bc2p, NCp, bc2, NC, 1, NC);

  GEMM(xl, wc2p, bc2p, outp, Mp, NCp, 512, 0, 0);
  copy2d<<<cdiv((long)Nn * NC, 256), 256, 0, stream>>>((float*)d_out, NC, outp, NCp, Nn, NC);

  #undef GEMM
}
